// FourierFFNBlock_71305047048415
// MI455X (gfx1250) — matrix operand order repaired and checked by reading, not compiled, not run on hardware
//
#include <hip/hip_runtime.h>
#include <hip/hip_bf16.h>
#include <math.h>

typedef __attribute__((ext_vector_type(16))) _Float16 v16h;
typedef __attribute__((ext_vector_type(8)))  _Float16 v8h;
typedef __attribute__((ext_vector_type(8)))  float    v8f;
typedef __attribute__((ext_vector_type(4)))  float    v4f;
typedef __attribute__((ext_vector_type(4)))  int      v4i;

#define N_FEAT 128
#define WAVES 4           // waves per block
#define ROWS_PER_WAVE 32
#define ROWS_PER_BLOCK (WAVES * ROWS_PER_WAVE)
#define BT_PAD 136        // f16 row stride: 272B -> 16B aligned, 4-bank rotation/row
#define XS_PAD 132        // f32 row stride: 528B -> 16B aligned, 4-bank rotation/row

#if __has_builtin(__builtin_amdgcn_global_load_async_to_lds_b128) && \
    __has_builtin(__builtin_amdgcn_s_wait_asynccnt)
#define USE_ASYNC_LDS 1
#else
#define USE_ASYNC_LDS 0
#endif

#define __AS1 __attribute__((address_space(1)))
#define __AS3 __attribute__((address_space(3)))

__device__ __forceinline__ float fast_tanh(float x) {
#if __has_builtin(__builtin_amdgcn_tanhf)
  return __builtin_amdgcn_tanhf(x);
#else
  float ax = __builtin_fabsf(x);
  float e  = __expf(-2.0f * ax);
  float t  = (1.0f - e) / (1.0f + e);
  return __builtin_copysignf(t, x);
#endif
}

// ---------------------------------------------------------------------------
// Precompute: BT[n][k] = dyt_w[k] * Re(ifft(b))[(n-k) mod 128]   (f16)
//             ybias[n] = sum_k dyt_b[k] * Re(ifft(b))[(n-k) mod 128]
// Re(ifft(b))[d] = (1/128) * sum_k ( br[k]*cos(2pi k d/128) - bi[k]*sin(2pi k d/128) )
// ---------------------------------------------------------------------------
__global__ __launch_bounds__(N_FEAT)
void fourier_precompute_kernel(const float* __restrict__ brl,
                               const float* __restrict__ bim,
                               const float* __restrict__ dytw,
                               const float* __restrict__ dytb,
                               _Float16* __restrict__ btw,   // [128*128]
                               float* __restrict__ ybias) {  // [128]
  __shared__ float cre[N_FEAT];
  const int t = threadIdx.x;            // 0..127
  const float two_pi_over_n = 6.2831853071795864769f / (float)N_FEAT;
  float s = 0.0f;
  for (int k = 0; k < N_FEAT; ++k) {
    float th = two_pi_over_n * (float)((k * t) & (N_FEAT - 1));
    s += brl[k] * __cosf(th) - bim[k] * __sinf(th);
  }
  cre[t] = s * (1.0f / (float)N_FEAT);
  __syncthreads();
  float yb = 0.0f;
  for (int k = 0; k < N_FEAT; ++k) {
    float c = cre[(t - k) & (N_FEAT - 1)];
    btw[t * N_FEAT + k] = (_Float16)(dytw[k] * c);
    yb += dytb[k] * c;
  }
  ybias[t] = yb;
}

// ---------------------------------------------------------------------------
// Main: out = x + tanh(alpha*x) @ B' + ybias, streaming 32 rows per wave.
// WMMA f32_16x16x32_f16, B' resident in LDS, x staged in LDS via async
// global->LDS loads (ASYNCcnt path) when the toolchain supports it.
// ---------------------------------------------------------------------------
__global__ __launch_bounds__(WAVES * 32)
void fourier_ffn_kernel(const float* __restrict__ x,
                        const float* __restrict__ alphap,
                        const _Float16* __restrict__ btw,
                        const float* __restrict__ ybias,
                        float* __restrict__ out,
                        int nrows) {
  __shared__ _Float16 bt[N_FEAT][BT_PAD];                 // 34816 B
  __shared__ float    yb_s[N_FEAT];                       //   512 B
  __shared__ float    xs[WAVES][ROWS_PER_WAVE][XS_PAD];   // 67584 B

  const int tid  = threadIdx.x;
  const int lane = tid & 31;
  const int w    = tid >> 5;
  const int half = lane >> 4;     // 0: lanes 0-15, 1: lanes 16-31
  const int mrow = lane & 15;
  const float alpha = alphap[0];

  const long rowbase = (long)blockIdx.x * ROWS_PER_BLOCK + (long)w * ROWS_PER_WAVE;
  const bool active = (rowbase + ROWS_PER_WAVE) <= (long)nrows;

  // ---- B' (32KB) -> LDS, ybias -> LDS ----
  for (int i = tid; i < (N_FEAT * N_FEAT) / 8; i += WAVES * 32) {
    int r = i >> 4;               // 16 8-elt chunks per 128-wide row
    int c = (i & 15) * 8;
#if USE_ASYNC_LDS
    __builtin_amdgcn_global_load_async_to_lds_b128(
        (__AS1 v4i*)(btw + (size_t)r * N_FEAT + c),
        (__AS3 v4i*)(&bt[r][c]), 0, 0);
#else
    float4 v = *(const float4*)(btw + (size_t)r * N_FEAT + c);
    *(float4*)(&bt[r][c]) = v;
#endif
  }
  yb_s[tid] = ybias[tid];

  // ---- stage this wave's 32x128 f32 x tile (16B per lane per row) ----
  if (active) {
#pragma unroll 4
    for (int r = 0; r < ROWS_PER_WAVE; ++r) {
#if USE_ASYNC_LDS
      __builtin_amdgcn_global_load_async_to_lds_b128(
          (__AS1 v4i*)(x + (size_t)(rowbase + r) * N_FEAT + lane * 4),
          (__AS3 v4i*)(&xs[w][r][lane * 4]), 0, 0);
#else
      v4f v = *(const v4f*)(x + (size_t)(rowbase + r) * N_FEAT + lane * 4);
      *(v4f*)(&xs[w][r][lane * 4]) = v;
#endif
    }
  }
#if USE_ASYNC_LDS
  __builtin_amdgcn_s_wait_asynccnt(0);   // this wave's async LDS writes done
#endif
  __syncthreads();                       // all waves' LDS writes visible
  if (!active) return;

  // ---- build A fragments: 2 M-tiles x 4 K-frags of 16x32 f16 ----
  // ISA layout: lane L holds row M=L&15; VGPR0-3: K = half*8+0..7, VGPR4-7: K = 16+half*8+0..7
  v16h afrag[2][4];
#pragma unroll
  for (int mt = 0; mt < 2; ++mt) {
    const float* xrow = &xs[w][mt * 16 + mrow][0];
#pragma unroll
    for (int kf = 0; kf < 4; ++kf) {
      const int k0 = kf * 32 + half * 8;
      v16h a;
#pragma unroll
      for (int i = 0; i < 8; ++i)
        a[i] = (_Float16)fast_tanh(alpha * xrow[k0 + i]);
#pragma unroll
      for (int i = 0; i < 8; ++i)
        a[8 + i] = (_Float16)fast_tanh(alpha * xrow[k0 + 16 + i]);
      afrag[mt][kf] = a;
    }
  }

  // ---- N-tiles: B frag from LDS (reused across both M-tiles), 4 K-step WMMA chain ----
#pragma unroll
  for (int nt = 0; nt < 8; ++nt) {
    const int n = nt * 16 + mrow;   // this lane's output column
    const float ybv = yb_s[n];
    v8f acc0 = {ybv, ybv, ybv, ybv, ybv, ybv, ybv, ybv};
    v8f acc1 = acc0;
#pragma unroll
    for (int kf = 0; kf < 4; ++kf) {
      // ISA B layout (32x16 f16): lane L holds col N=L&15; VGPR0-3: K = half*8+0..7, VGPR4-7: K = 16+half*8+0..7
      const int k0 = kf * 32 + half * 8;
      union { v16h v; struct { v8h lo, hi; } s; } ub;
      ub.s.lo = *(const v8h*)(&bt[n][k0]);
      ub.s.hi = *(const v8h*)(&bt[n][k0 + 16]);
      acc0 = __builtin_amdgcn_wmma_f32_16x16x32_f16(
          false, afrag[0][kf], false, ub.v, (short)0, acc0, false, false);
      acc1 = __builtin_amdgcn_wmma_f32_16x16x32_f16(
          false, afrag[1][kf], false, ub.v, (short)0, acc1, false, false);
    }
    // ---- residual add + store. D layout: VGPR g -> row g + half*8, col = n ----
#pragma unroll
    for (int g = 0; g < 8; ++g) {
      const int rA = g + half * 8;
      float r0 = acc0[g] + xs[w][rA][n];
      out[(size_t)(rowbase + rA) * N_FEAT + n] = r0;
      float r1 = acc1[g] + xs[w][16 + rA][n];
      out[(size_t)(rowbase + 16 + rA) * N_FEAT + n] = r1;
    }
  }
}

// ---------------------------------------------------------------------------
extern "C" void kernel_launch(void* const* d_in, const int* in_sizes, int n_in,
                              void* d_out, int out_size, void* d_ws, size_t ws_size,
                              hipStream_t stream) {
  // setup_inputs order: x, alpha, dyt_weight, dyt_bias, a_real, a_imag,
  //                     b_real, b_imag, ffn_bias   (a_*, ffn_bias are dead code)
  const float* x     = (const float*)d_in[0];
  const float* alpha = (const float*)d_in[1];
  const float* dytw  = (const float*)d_in[2];
  const float* dytb  = (const float*)d_in[3];
  const float* brl   = (const float*)d_in[6];
  const float* bim   = (const float*)d_in[7];
  float* out = (float*)d_out;

  _Float16* btw  = (_Float16*)d_ws;
  float*    ybia = (float*)((char*)d_ws + (size_t)N_FEAT * N_FEAT * sizeof(_Float16));

  const int nrows = in_sizes[0] / N_FEAT;  // 32*8192 = 262144

  fourier_precompute_kernel<<<1, N_FEAT, 0, stream>>>(brl, bim, dytw, dytb, btw, ybia);

  const int grid = (nrows + ROWS_PER_BLOCK - 1) / ROWS_PER_BLOCK;  // 2048
  fourier_ffn_kernel<<<grid, WAVES * 32, 0, stream>>>(x, alpha, btw, ybia, out, nrows);
}
